// MIM_Model_65017214927480
// MI455X (gfx1250) — compile-verified, matrix operand order checked by reading, not run on hardware
//
#include <hip/hip_runtime.h>
#include <cstddef>

// ---------------- problem constants ----------------
#define BB    4
#define T_IN  10
#define T_OUT 10
#define HP    32
#define WP    32
#define FC    16
#define NH    64
#define STEPS (T_IN + T_OUT - 1)   // 19
#define HW    (HP * WP)            // 1024
#define NTOT  (BB * HW)            // 4096 spatial columns (GEMM N)
#define S64   (NH * NTOT)          // one 64-channel plane, elements

typedef __attribute__((ext_vector_type(16))) __bf16 v16bf;
typedef __attribute__((ext_vector_type(8)))  __bf16 bf16x8;
typedef __attribute__((ext_vector_type(8)))  float  v8f;

__device__ __forceinline__ float sigf(float x) { return 1.0f / (1.0f + __expf(-x)); }

static inline int kpad32(int k) { return (k + 31) & ~31; }

// ======================================================================
// WMMA bf16 GEMM:  C[M][4096] (f32) = A[M][Kp] (bf16) x Bt[4096][Kp] (bf16)
//  - A is zero-padded to Kp (multiple of 32) at weight-conversion time
//  - B is produced K-transposed (n-major) by im2col/cvt, so both LDS tiles
//    are staged with pure 16B vector copies and fragments are contiguous
//  - block tile 64(M) x 128(N), 8 wave32 (2x4), double-buffered LDS
// ======================================================================
__device__ __forceinline__ void stage_tiles(const __bf16* __restrict__ A,
                                            const __bf16* __restrict__ Bt,
                                            int M, int Kp, int m0, int n0, int k0,
                                            __bf16 (*As)[32], __bf16 (*Bs)[32], int tid)
{
    // A tile: 64 rows x 32 k, one 16B vector per thread
    {
        const int row = tid >> 2;          // 0..63
        const int grp = tid & 3;           // 0..3
        bf16x8 v = {};
        const __bf16* src = A + (size_t)(m0 + row) * Kp + k0 + grp * 8;
        if ((m0 + row) < M) {
            v = *(const bf16x8*)src;
            if (k0 + 32 < Kp) __builtin_prefetch(src + 32, 0, 1);   // global_prefetch_b8
        }
        *(bf16x8*)&As[row][grp * 8] = v;
    }
    // B tile: 128 cols x 32 k (already transposed in global), two 16B vectors per thread
    {
        const int nl   = tid >> 1;         // 0..127
        const int half = tid & 1;          // 0..1
        const __bf16* src = Bt + (size_t)(n0 + nl) * Kp + k0 + half * 16;
        bf16x8 v0 = *(const bf16x8*)src;
        bf16x8 v1 = *(const bf16x8*)(src + 8);
        *(bf16x8*)&Bs[nl][half * 16]     = v0;
        *(bf16x8*)&Bs[nl][half * 16 + 8] = v1;
        if (k0 + 32 < Kp) __builtin_prefetch(src + 32, 0, 1);
    }
}

__device__ __forceinline__ void store_tile(float* __restrict__ C, const v8f& a,
                                           int M, int row0, int col)
{
#pragma unroll
    for (int v = 0; v < 8; ++v) {
        const int rr = row0 + v;                      // VGPR v -> M = v (+8 for hi lanes)
        if (rr < M) C[(size_t)rr * NTOT + col] = a[v];
    }
}

__global__ void __launch_bounds__(256)
k_gemm(const __bf16* __restrict__ A, const __bf16* __restrict__ Bt,
       float* __restrict__ C, int M, int Kp)
{
    __shared__ alignas(64) __bf16 As[2][64][32];
    __shared__ alignas(64) __bf16 Bs[2][128][32];

    const int tid  = threadIdx.x;
    const int n0   = blockIdx.x * 128;
    const int m0   = blockIdx.y * 64;
    const int lane = tid & 31;
    const int wave = tid >> 5;
    const int wm   = wave >> 2;   // 0..1 : 32-row strip
    const int wn   = wave & 3;    // 0..3 : 32-col strip

    v8f acc00 = {}, acc01 = {}, acc10 = {}, acc11 = {};

    const int nch = Kp >> 5;
    stage_tiles(A, Bt, M, Kp, m0, n0, 0, As[0], Bs[0], tid);
    __syncthreads();

    for (int ch = 0; ch < nch; ++ch) {
        const int cur = ch & 1;
        if (ch + 1 < nch)
            stage_tiles(A, Bt, M, Kp, m0, n0, (ch + 1) * 32, As[cur ^ 1], Bs[cur ^ 1], tid);

        const int r  = lane & 15;            // row (A) / col (B) within 16
        const int hf = (lane >> 4) << 4;     // K half: lanes 16-31 take K=16..31
        v16bf a0 = *(const v16bf*)&As[cur][wm * 32 + r][hf];
        v16bf a1 = *(const v16bf*)&As[cur][wm * 32 + 16 + r][hf];
        v16bf b0 = *(const v16bf*)&Bs[cur][wn * 32 + r][hf];
        v16bf b1 = *(const v16bf*)&Bs[cur][wn * 32 + 16 + r][hf];

        acc00 = __builtin_amdgcn_wmma_f32_16x16x32_bf16(false, a0, false, b0, (short)0, acc00, false, false);
        acc01 = __builtin_amdgcn_wmma_f32_16x16x32_bf16(false, a0, false, b1, (short)0, acc01, false, false);
        acc10 = __builtin_amdgcn_wmma_f32_16x16x32_bf16(false, a1, false, b0, (short)0, acc10, false, false);
        acc11 = __builtin_amdgcn_wmma_f32_16x16x32_bf16(false, a1, false, b1, (short)0, acc11, false, false);
        __syncthreads();
    }

    const int r    = lane & 15;
    const int hi   = (lane >> 4) << 3;       // hi lanes hold M = v+8
    const int colb = n0 + wn * 32 + r;
    const int rowb = m0 + wm * 32 + hi;
    store_tile(C, acc00, M, rowb,      colb);
    store_tile(C, acc01, M, rowb,      colb + 16);
    store_tile(C, acc10, M, rowb + 16, colb);
    store_tile(C, acc11, M, rowb + 16, colb + 16);
}

// ======================================================================
// Support kernels; activations laid out [C][B][HW]: idx = c*4096 + b*1024 + hw
// ======================================================================
__global__ void k_cvt_pad(const float* __restrict__ s, __bf16* __restrict__ d,
                          int K, int Kp)          // weights: [M][K] -> [M][Kp] zero padded
{
    int i = blockIdx.x * 256 + threadIdx.x;       // over M*Kp
    int m = i / Kp, k = i % Kp;
    d[i] = (__bf16)((k < K) ? s[(size_t)m * K + k] : 0.0f);
}

// activation -> transposed bf16 B-matrix for 1x1 convs: abf[n][c] = x[c][n]
__global__ void k_cvt_tr(const float* __restrict__ x, __bf16* __restrict__ bt, int Cin)
{
    int i = blockIdx.x * 256 + threadIdx.x;       // over NTOT*Cin
    int n = i / Cin, ch = i % Cin;
    bt[i] = (__bf16)x[(size_t)ch * NTOT + n];
}

__global__ void k_zero(float* __restrict__ p)
{
    p[blockIdx.x * 256 + threadIdx.x] = 0.0f;
}

__global__ void k_copy(const float* __restrict__ s, float* __restrict__ d)
{
    int i = blockIdx.x * 256 + threadIdx.x;
    d[i] = s[i];
}

__global__ void k_sub(const float* __restrict__ a, const float* __restrict__ b, float* __restrict__ d)
{
    int i = blockIdx.x * 256 + threadIdx.x;
    d[i] = a[i] - b[i];
}

// im2col for 5x5 / pad 2, emitted TRANSPOSED + K-padded: Bt[n][k], k in [0,Kp)
__global__ void k_im2col(const float* __restrict__ x, __bf16* __restrict__ Bt, int K, int Kp)
{
    int i = blockIdx.x * 256 + threadIdx.x;       // over NTOT*Kp, consecutive threads = consecutive k
    int n = i / Kp, k = i % Kp;
    float v = 0.0f;
    if (k < K) {
        int c = k / 25, r = k % 25;
        int dy = r / 5 - 2, dx = r % 5 - 2;
        int b = n >> 10, hw = n & 1023;
        int y = (hw >> 5) + dy, xx = (hw & 31) + dx;
        if ((unsigned)y < 32u && (unsigned)xx < 32u)
            v = x[(size_t)c * NTOT + b * HW + y * 32 + xx];
    }
    Bt[i] = (__bf16)v;
}

// LayerNorm over (C,H,W) per batch: pass 1 stats, pass 2 normalize in place
__global__ void k_ln_stats(const float* __restrict__ x, int C, float* __restrict__ mv)
{
    const int b = blockIdx.x, tid = threadIdx.x;
    const int n = C * HW;
    float s = 0.f, ss = 0.f;
    for (int i = tid; i < n; i += 256) {
        int c = i >> 10, hw = i & 1023;
        float v = x[(size_t)c * NTOT + b * HW + hw];
        s += v; ss += v * v;
    }
    __shared__ float rs[256], rq[256];
    rs[tid] = s; rq[tid] = ss; __syncthreads();
    for (int o = 128; o > 0; o >>= 1) {
        if (tid < o) { rs[tid] += rs[tid + o]; rq[tid] += rq[tid + o]; }
        __syncthreads();
    }
    if (tid == 0) {
        float m = rs[0] / n;
        float var = rq[0] / n - m * m;
        mv[b * 2] = m;
        mv[b * 2 + 1] = rsqrtf(var + 1e-5f);
    }
}

__global__ void k_ln_apply(float* __restrict__ x, const float* __restrict__ mv)
{
    int i = blockIdx.x * 256 + threadIdx.x;
    int b = (i >> 10) & 3;
    x[i] = (x[i] - mv[b * 2]) * mv[b * 2 + 1];
}

// net = t < T_IN ? frame : mask*frame + (1-mask)*x_gen     (also NHWC->C-major)
__global__ void k_make_net(const float* __restrict__ frames, const float* __restrict__ mask,
                           const float* __restrict__ xgen, float* __restrict__ net, int t)
{
    int i = blockIdx.x * 256 + threadIdx.x;   // 16*4096
    int c = i >> 12, n = i & 4095, b = n >> 10, hw = n & 1023;
    float f = frames[(((size_t)b * (T_IN + T_OUT) + t) * HW + hw) * FC + c];
    float v = f;
    if (t >= T_IN) {
        float mk = mask[(((size_t)b * (T_OUT - 1) + (t - T_IN)) * HW + hw) * FC + c];
        v = mk * f + (1.0f - mk) * xgen[i];
    }
    net[i] = v;
}

// ST-LSTM gates: gx=ln(conv(x,wx)) 7ch-blocks, gh 4, gm 3 -> c,m updated + mem=concat
__global__ void k_stlstm_gates(const float* __restrict__ gx, const float* __restrict__ gh,
                               const float* __restrict__ gm, float* __restrict__ c,
                               float* __restrict__ m, float* __restrict__ mem)
{
    int i = blockIdx.x * 256 + threadIdx.x;
    const int S = S64;
    float cn = sigf(gx[S + i] + gh[S + i] + 1.0f) * c[i]
             + sigf(gx[i] + gh[i]) * tanhf(gx[2 * S + i] + gh[2 * S + i]);
    float mn = sigf(gx[4 * S + i] + gm[S + i] + 1.0f) * m[i]
             + sigf(gx[3 * S + i] + gm[i]) * tanhf(gx[5 * S + i] + gm[2 * S + i]);
    c[i] = cn; m[i] = mn; mem[i] = cn; mem[S + i] = mn;
}

__global__ void k_stlstm_out(const float* __restrict__ gx, const float* __restrict__ gh,
                             const float* __restrict__ go, const float* __restrict__ glast,
                             float* __restrict__ h)
{
    int i = blockIdx.x * 256 + threadIdx.x;
    const int S = S64;
    h[i] = sigf(gx[6 * S + i] + gh[3 * S + i] + go[i]) * tanhf(glast[i]);
}

// MIMN cell gates: gh/gxx = ln(conv) with [i,g,f,o] blocks; ctw[128][HW], ocw[64][HW]
__global__ void k_mimn_gates(const float* __restrict__ gh, const float* __restrict__ gxx,
                             const float* __restrict__ cin, const float* __restrict__ ctw,
                             const float* __restrict__ ocw, float* __restrict__ hout,
                             float* __restrict__ cout, int zero_out)
{
    int i = blockIdx.x * 256 + threadIdx.x;
    const int S = S64;
    int ch = i >> 12, hw = i & 1023;
    float cv = cin[i];
    float ic = cv * ctw[ch * HW + hw];
    float fc = cv * ctw[(ch + NH) * HW + hw];
    float ii = sigf(gh[i] + ic + gxx[i]);
    float ff = sigf(gh[2 * S + i] + fc + gxx[2 * S + i] + 1.0f);
    float cn = ff * cv + ii * tanhf(gh[S + i] + gxx[S + i]);
    float hn = sigf(gh[3 * S + i] + gxx[3 * S + i] + cn * ocw[ch * HW + hw]) * tanhf(cn);
    if (zero_out) { hn = 0.0f; cn = 0.0f; }
    hout[i] = hn; cout[i] = cn;
}

// MIM block post: gt=[i,g,o], gs=[i,g,f,o], gx7=[i,g,f,o,ip,gp,fp]
__global__ void k_mim_post(const float* __restrict__ gt, const float* __restrict__ gs,
                           const float* __restrict__ gx7, const float* __restrict__ cmid,
                           float* __restrict__ ci, float* __restrict__ m, float* __restrict__ mem)
{
    int i = blockIdx.x * 256 + threadIdx.x;
    const int S = S64;
    float ii = sigf(gx7[i] + gt[i]);
    float gg = tanhf(gx7[S + i] + gt[S + i]);
    float mn = sigf(gx7[6 * S + i] + gs[2 * S + i] + 1.0f) * m[i]
             + sigf(gx7[4 * S + i] + gs[i]) * tanhf(gx7[5 * S + i] + gs[S + i]);
    float cn = cmid[i] + ii * gg;
    ci[i] = cn; m[i] = mn; mem[i] = cn; mem[S + i] = mn;
}

__global__ void k_mim_out(const float* __restrict__ gt, const float* __restrict__ gs,
                          const float* __restrict__ gx7, const float* __restrict__ glast,
                          float* __restrict__ h)
{
    int i = blockIdx.x * 256 + threadIdx.x;
    const int S = S64;
    h[i] = sigf(gx7[3 * S + i] + gt[2 * S + i] + gs[3 * S + i]) * tanhf(glast[i]);
}

// write next_frames (C-major -> NHWC) + per-block squared-error partials
__global__ void k_store_out(const float* __restrict__ pred, const float* __restrict__ frames,
                            float* __restrict__ outf, float* __restrict__ part, int t)
{
    int i = blockIdx.x * 256 + threadIdx.x;   // 16*4096
    int c = i >> 12, n = i & 4095, b = n >> 10, hw = n & 1023;
    float p = pred[i];
    outf[(((size_t)b * STEPS + t) * HW + hw) * FC + c] = p;
    float tgt = frames[(((size_t)b * (T_IN + T_OUT) + (t + 1)) * HW + hw) * FC + c];
    float d = p - tgt;
    __shared__ float red[256];
    red[threadIdx.x] = d * d; __syncthreads();
    for (int o = 128; o > 0; o >>= 1) {
        if (threadIdx.x < o) red[threadIdx.x] += red[threadIdx.x + o];
        __syncthreads();
    }
    if (threadIdx.x == 0) part[blockIdx.x] = red[0];
}

__global__ void k_loss_final(const float* __restrict__ part, float* __restrict__ out)
{
    __shared__ float red[256];
    float s = 0.0f;
    for (int i = threadIdx.x; i < STEPS * 256; i += 256) s += part[i];
    red[threadIdx.x] = s; __syncthreads();
    for (int o = 128; o > 0; o >>= 1) {
        if (threadIdx.x < o) red[threadIdx.x] += red[threadIdx.x + o];
        __syncthreads();
    }
    if (threadIdx.x == 0) out[0] = red[0] / (float)((size_t)BB * STEPS * HW * FC);
}

// ======================================================================
// Host orchestration
// ======================================================================
extern "C" void kernel_launch(void* const* d_in, const int* in_sizes, int n_in,
                              void* d_out, int out_size, void* d_ws, size_t ws_size,
                              hipStream_t stream)
{
    (void)in_sizes; (void)n_in; (void)out_size; (void)ws_size;

    const float* frames = (const float*)d_in[0];
    const float* maskp  = (const float*)d_in[1];
    // params flattened in dict-insertion / list order:
    // stl0{wx,wh,wm,wo,wlast}=2..6, mim[i]{wt,ws,wx,wh_m,wx_m,ctw,ocw,wlast}=7+8i,
    // mimn[i]{wh,wx,ctw,ocw}=31+4i, conv_last=43
    const float* p_s_wx    = (const float*)d_in[2];
    const float* p_s_wh    = (const float*)d_in[3];
    const float* p_s_wm    = (const float*)d_in[4];
    const float* p_s_wo    = (const float*)d_in[5];
    const float* p_s_wlast = (const float*)d_in[6];
    const float* p_mim[3][8];
    for (int i = 0; i < 3; ++i)
        for (int j = 0; j < 8; ++j) p_mim[i][j] = (const float*)d_in[7 + i * 8 + j];
    const float* p_mimn[3][4];
    for (int i = 0; i < 3; ++i)
        for (int j = 0; j < 4; ++j) p_mimn[i][j] = (const float*)d_in[31 + i * 4 + j];
    const float* p_convlast = (const float*)d_in[43];

    // ---- workspace bump allocator (256B aligned) ----
    char*  base = (char*)d_ws;
    size_t off  = 0;
    auto alloc = [&](size_t bytes) -> void* {
        void* p = base + off;
        off += (bytes + 255) & ~(size_t)255;
        return p;
    };

    // contiguous zero-init state block
    float* states = (float*)alloc((size_t)(18 * S64 + FC * NTOT) * 4);
    float *h[4], *c[4], *dh[3], *dc[3], *mc[3], *stm, *xgen;
    {
        float* p = states;
        for (int i = 0; i < 4; ++i) { h[i] = p; p += S64; }
        for (int i = 0; i < 4; ++i) { c[i] = p; p += S64; }
        stm = p; p += S64;
        for (int i = 0; i < 3; ++i) { dh[i] = p; p += S64; }
        for (int i = 0; i < 3; ++i) { dc[i] = p; p += S64; }
        for (int i = 0; i < 3; ++i) { mc[i] = p; p += S64; }
        xgen = p;
    }
    float* preh  = (float*)alloc((size_t)S64 * 4);
    float* dtmp  = (float*)alloc((size_t)S64 * 4);
    float* net   = (float*)alloc((size_t)FC * NTOT * 4);
    float* mem   = (float*)alloc((size_t)2 * S64 * 4);           // 128 ch
    float* cmid  = (float*)alloc((size_t)S64 * 4);
    float* gx    = (float*)alloc((size_t)7 * S64 * 4);           // up to 448 ch
    float* gh    = (float*)alloc((size_t)4 * S64 * 4);           // up to 256 ch
    float* gs    = (float*)alloc((size_t)4 * S64 * 4);
    float* gt    = (float*)alloc((size_t)3 * S64 * 4);
    float* go    = (float*)alloc((size_t)S64 * 4);
    float* glast = (float*)alloc((size_t)S64 * 4);
    float* ghm   = (float*)alloc((size_t)4 * S64 * 4);
    float* gxm   = (float*)alloc((size_t)4 * S64 * 4);
    __bf16* im2c = (__bf16*)alloc((size_t)NTOT * 3200 * 2);      // max Kp=3200 (transposed)
    __bf16* abf  = (__bf16*)alloc((size_t)NTOT * 128 * 2);       // 1x1 conv input (transposed)
    float* lnst  = (float*)alloc(8 * 4);
    float* lpart = (float*)alloc((size_t)STEPS * 256 * 4);

    // bf16 weight mirrors, zero-padded to Kp: A matrices [Cout][Kp]
    auto mkbf = [&](const float* src, int M, int K) -> __bf16* {
        const int Kp = kpad32(K);
        __bf16* d = (__bf16*)alloc((size_t)M * Kp * sizeof(__bf16));
        k_cvt_pad<<<((size_t)M * Kp) / 256, 256, 0, stream>>>(src, d, K, Kp);
        return d;
    };

    __bf16* bs_wx    = mkbf(p_s_wx,    7 * NH, FC * 25);
    __bf16* bs_wh    = mkbf(p_s_wh,    4 * NH, NH * 25);
    __bf16* bs_wm    = mkbf(p_s_wm,    3 * NH, NH * 25);
    __bf16* bs_wo    = mkbf(p_s_wo,    NH,     2 * NH * 25);
    __bf16* bs_wlast = mkbf(p_s_wlast, NH,     2 * NH);
    __bf16 *bm_wt[3], *bm_ws[3], *bm_wx[3], *bm_whm[3], *bm_wxm[3], *bm_wlast[3];
    __bf16 *bn_wh[3], *bn_wx[3];
    for (int i = 0; i < 3; ++i) {
        bm_wt[i]    = mkbf(p_mim[i][0], 3 * NH, NH * 25);
        bm_ws[i]    = mkbf(p_mim[i][1], 4 * NH, NH * 25);
        bm_wx[i]    = mkbf(p_mim[i][2], 7 * NH, NH * 25);
        bm_whm[i]   = mkbf(p_mim[i][3], 4 * NH, NH * 25);
        bm_wxm[i]   = mkbf(p_mim[i][4], 4 * NH, NH * 25);
        bm_wlast[i] = mkbf(p_mim[i][7], NH,     2 * NH);
        bn_wh[i]    = mkbf(p_mimn[i][0], 4 * NH, NH * 25);
        bn_wx[i]    = mkbf(p_mimn[i][1], 4 * NH, NH * 25);
    }
    __bf16* b_convlast = mkbf(p_convlast, FC, NH);

    // zero recurrent state (every call: deterministic)
    k_zero<<<(18 * S64 + FC * NTOT) / 256, 256, 0, stream>>>(states);

    const int G64 = S64 / 256;               // 1024 blocks for a 64ch plane
    auto conv5 = [&](const float* x, int Cin, const __bf16* Wb, int Cout, float* outp) {
        const int K = Cin * 25, Kp = kpad32(K);
        k_im2col<<<((size_t)NTOT * Kp) / 256, 256, 0, stream>>>(x, im2c, K, Kp);
        k_gemm<<<dim3(NTOT / 128, (Cout + 63) / 64), 256, 0, stream>>>(Wb, im2c, outp, Cout, Kp);
    };
    auto conv1 = [&](const float* x, int Cin, const __bf16* Wb, int Cout, float* outp) {
        k_cvt_tr<<<((size_t)NTOT * Cin) / 256, 256, 0, stream>>>(x, abf, Cin);
        k_gemm<<<dim3(NTOT / 128, (Cout + 63) / 64), 256, 0, stream>>>(Wb, abf, outp, Cout, Cin);
    };
    auto lnorm = [&](float* x, int C) {
        k_ln_stats<<<BB, 256, 0, stream>>>(x, C, lnst);
        k_ln_apply<<<(C * NTOT) / 256, 256, 0, stream>>>(x, lnst);
    };

    float* outf  = (float*)d_out;
    float* lossp = outf + (size_t)BB * STEPS * HW * FC;

    for (int t = 0; t < STEPS; ++t) {
        k_make_net<<<(FC * NTOT) / 256, 256, 0, stream>>>(frames, maskp, xgen, net, t);
        k_copy<<<G64, 256, 0, stream>>>(h[0], preh);

        // ---- ST-LSTM layer 0 ----
        conv5(net,  FC, bs_wx, 7 * NH, gx); lnorm(gx, 7 * NH);
        conv5(h[0], NH, bs_wh, 4 * NH, gh); lnorm(gh, 4 * NH);
        conv5(stm,  NH, bs_wm, 3 * NH, gs); lnorm(gs, 3 * NH);
        k_stlstm_gates<<<G64, 256, 0, stream>>>(gx, gh, gs, c[0], stm, mem);
        conv5(mem, 2 * NH, bs_wo, NH, go); lnorm(go, NH);
        conv1(mem, 2 * NH, bs_wlast, NH, glast);
        k_stlstm_out<<<G64, 256, 0, stream>>>(gx, gh, go, glast, h[0]);

        k_sub<<<G64, 256, 0, stream>>>(h[0], preh, dtmp);

        // ---- MIM layers 1..3 ----
        for (int i = 1; i < 4; ++i) {
            const int li = i - 1;
            const float* din = (i == 1) ? dtmp : dh[i - 2];

            // diff cell (MIMN): x=din, h=dh[li], c=dc[li]
            conv5(dh[li], NH, bn_wh[li], 4 * NH, ghm); lnorm(ghm, 4 * NH);
            conv5(din,    NH, bn_wx[li], 4 * NH, gxm); lnorm(gxm, 4 * NH);
            k_mimn_gates<<<G64, 256, 0, stream>>>(ghm, gxm, dc[li], p_mimn[li][2], p_mimn[li][3],
                                                  dh[li], dc[li], (t == 0) ? 1 : 0);

            // MIM block: x=h[i-1], diff_h=dh[li], h=h[i], c=c[i], m=stm, mims_c=mc[li]
            conv5(h[i],     NH, bm_wt[li], 3 * NH, gt); lnorm(gt, 3 * NH);
            conv5(stm,      NH, bm_ws[li], 4 * NH, gs); lnorm(gs, 4 * NH);
            conv5(h[i - 1], NH, bm_wx[li], 7 * NH, gx); lnorm(gx, 7 * NH);
            // internal MIMS: x=dh[li], h=c[i], c=mc[li] -> c_mid, mc
            conv5(c[i],   NH, bm_whm[li], 4 * NH, ghm); lnorm(ghm, 4 * NH);
            conv5(dh[li], NH, bm_wxm[li], 4 * NH, gxm); lnorm(gxm, 4 * NH);
            k_mimn_gates<<<G64, 256, 0, stream>>>(ghm, gxm, mc[li], p_mim[li][5], p_mim[li][6],
                                                  cmid, mc[li], 0);
            k_mim_post<<<G64, 256, 0, stream>>>(gt, gs, gx, cmid, c[i], stm, mem);
            conv1(mem, 2 * NH, bm_wlast[li], NH, glast);
            k_mim_out<<<G64, 256, 0, stream>>>(gt, gs, gx, glast, h[i]);
        }

        // ---- x_gen + output + loss partials ----
        conv1(h[3], NH, b_convlast, FC, xgen);
        k_store_out<<<(FC * NTOT) / 256, 256, 0, stream>>>(xgen, frames, outf, lpart + t * 256, t);
    }

    k_loss_final<<<1, 256, 0, stream>>>(lpart, lossp);
}
